// SpatiallyVaryingPhysicalLayer_16561393893677
// MI455X (gfx1250) — matrix operand order repaired and checked by reading, not compiled, not run on hardware
//
#include <hip/hip_runtime.h>

typedef __attribute__((ext_vector_type(16))) _Float16 v16h;
typedef __attribute__((ext_vector_type(8)))  _Float16 v8h;
typedef __attribute__((ext_vector_type(4)))  _Float16 v4h;
typedef __attribute__((ext_vector_type(8)))  float    v8f;
typedef __attribute__((ext_vector_type(4)))  float    v4f;
typedef __attribute__((ext_vector_type(4)))  unsigned int v4u;
typedef __attribute__((ext_vector_type(8)))  int      v8i;
typedef __attribute__((ext_vector_type(4)))  int      v4i;

#define HIDN 64
#define ZK   15
#define KS   31
#define PP   128
#define SS   64
#define NHW  11          // patch grid is 11x11
#define NPB  121         // patches per batch image
#define NB   2
#define NC   3
#define NTOT (NB*NPB)    // 242
#define HW   768

// LDS geometry for the conv kernel (f16 halves unless noted)
#define PROWS 158        // 128 + 2*15 row halo
#define PCOLS 176        // 16 left halo + 128 + right zero pad (A reads up to col 175)
#define HR 15
#define HC 16
#define PATCH_HALVES (PROWS*PCOLS)              // 27808 halves = 55616 B
#define TOEP_HALVES  (31*64*16)                 // 31744 halves = 63488 B
#define LDS_HANN_OFF_BYTES ((PATCH_HALVES+TOEP_HALVES)*2)   // 119104 B (16B aligned)
#define LDS_TOTAL_BYTES (LDS_HANN_OFF_BYTES + 128*4)        // 119616 B

#if __has_builtin(__builtin_amdgcn_tensor_load_to_lds) && \
    __has_builtin(__builtin_amdgcn_s_wait_tensorcnt)
#define HAVE_TDM 1
#else
#define HAVE_TDM 0
#endif

// ---------------------------------------------------------------- zero output
extern "C" __global__ void sv_zero(float* __restrict__ out, int ntot) {
  int i = blockIdx.x * blockDim.x + threadIdx.x;
  int stride = gridDim.x * blockDim.x;
  for (; i < ntot; i += stride) out[i] = 0.0f;
}

// ------------------------------------------- cast full image f32 -> f16 (once)
extern "C" __global__ void sv_cast(const float* __restrict__ x,
                                   _Float16* __restrict__ xh, int nvec4) {
  int i = blockIdx.x * blockDim.x + threadIdx.x;
  int stride = gridDim.x * blockDim.x;
  for (; i < nvec4; i += stride) {
    v4f f = *(const v4f*)(x + (size_t)i * 4);
    v4h h;
    h[0] = (_Float16)f[0]; h[1] = (_Float16)f[1];
    h[2] = (_Float16)f[2]; h[3] = (_Float16)f[3];
    *(v4h*)(xh + (size_t)i * 4) = h;
  }
}

// ------------------------------------------------- tiny MLP: coords -> coeffs
extern "C" __global__ void sv_coeffs(const float* __restrict__ W1,
                                     const float* __restrict__ b1,
                                     const float* __restrict__ W2,
                                     const float* __restrict__ b2,
                                     float* __restrict__ coeffs) {
  int nidx = blockIdx.x * blockDim.x + threadIdx.x;
  if (nidx >= NTOT) return;
  int t  = nidx % NPB;
  int ih = t / NHW, iw = t % NHW;
  float gy = ((float)(ih * SS) + 64.0f) / 768.0f * 2.0f - 1.0f;
  float gx = ((float)(iw * SS) + 64.0f) / 768.0f * 2.0f - 1.0f;
  float h[HIDN];
#pragma unroll 8
  for (int k = 0; k < HIDN; ++k)
    h[k] = tanhf(gy * W1[k] + gx * W1[HIDN + k] + b1[k]);
  for (int z = 0; z < ZK; ++z) {
    float s = b2[z];
#pragma unroll 8
    for (int k = 0; k < HIDN; ++k) s += h[k] * W2[k * ZK + z];
    coeffs[nidx * 16 + z] = s;
  }
}

// -------------------------- kern = normalize(softplus(coeff . basis)), per (n,c)
extern "C" __global__ void __launch_bounds__(256)
sv_kern(const float* __restrict__ coeffs, const float* __restrict__ basis,
        float* __restrict__ kern) {
  __shared__ float red[256];
  __shared__ float cz[16];
  const int tid = threadIdx.x;
  const int wg  = blockIdx.x;            // 0..725
  const int n   = wg / NC, c = wg % NC;
  if (tid < ZK) cz[tid] = coeffs[n * 16 + tid];
  __syncthreads();

  float vals[4];
  float lsum = 0.0f;
#pragma unroll
  for (int u = 0; u < 4; ++u) {
    int e = tid + u * 256;
    float v = 0.0f;
    if (e < KS * KS) {
      float s = 0.0f;
#pragma unroll
      for (int z = 0; z < ZK; ++z)
        s += cz[z] * basis[((size_t)z * NC + c) * (KS * KS) + e];
      v = fmaxf(s, 0.0f) + log1pf(expf(-fabsf(s)));   // stable softplus
      lsum += v;
    }
    vals[u] = v;
  }
  red[tid] = lsum;
  __syncthreads();
  for (int s2 = 128; s2 > 0; s2 >>= 1) {
    if (tid < s2) red[tid] += red[tid + s2];
    __syncthreads();
  }
  const float inv = 1.0f / (red[0] + 1e-8f);
  float* kout = kern + (size_t)(n * NC + c) * (KS * KS);
#pragma unroll
  for (int u = 0; u < 4; ++u) {
    int e = tid + u * 256;
    if (e < KS * KS) kout[e] = vals[u] * inv;
  }
}

// ------------------------------ depthwise 31x31 "same" conv via Toeplitz WMMA
extern "C" __global__ void __launch_bounds__(256)
sv_conv(const _Float16* __restrict__ xh, const float* __restrict__ kern,
        float* __restrict__ out) {
  extern __shared__ __align__(16) char lds_raw[];
  _Float16* PATCH = (_Float16*)lds_raw;                       // [158][176]
  _Float16* TOEP  = (_Float16*)lds_raw + PATCH_HALVES;        // [31][64][16]
  float*    HANN  = (float*)(lds_raw + LDS_HANN_OFF_BYTES);   // [128]

  const int tid = threadIdx.x;
  const int wg  = blockIdx.x;            // 0..725 => (patch n, channel c)
  const int n   = wg / NC;
  const int c   = wg % NC;
  const int b   = n / NPB;
  const int t   = n % NPB;
  const int ih  = t / NHW, iw = t % NHW;
  const int r0  = ih * SS, c0 = iw * SS;
  const int plane = b * NC + c;
  const float* kptr = kern + (size_t)(n * NC + c) * (KS * KS);
  const _Float16* src = xh + ((size_t)plane * HW + r0) * HW + c0;

  // warm L2 for this block's kernel data (gfx1250 global_prefetch_b8)
  if (tid < 8) __builtin_prefetch(kptr + tid * 128, 0, 1);

  // ---- phase 0: zero the haloed patch, build Hann table, build Toeplitz mats
  {
    unsigned int* p32 = (unsigned int*)PATCH;
    for (int i = tid; i < PATCH_HALVES / 2; i += 256) p32[i] = 0u;
    if (tid < PP)
      HANN[tid] = 0.5f * (1.0f - __cosf(6.28318530717958647692f * (float)tid / 128.0f));
  }
  // 31 banded Toeplitz B matrices (64x16 f16 each):
  // out(y,x) = sum_{ky=0..30, kx=1..31} kern(30-ky, 31-kx) * padded(y+ky, x+kx)
  for (int e = tid; e < 31 * 64 * 16; e += 256) {
    int ky = e >> 10;
    int j  = (e >> 4) & 63;
    int xl = e & 15;
    int kx = j - xl;
    float v = 0.0f;
    if (kx >= 1 && kx <= 31) v = kptr[(30 - ky) * KS + (31 - kx)];
    TOEP[ky * 1024 + j * 16 + xl] = (_Float16)v;
  }
  __syncthreads();

  // ---- phase 1: DMA the 128x128 f16 tile into the zeroed halo buffer.
  // TDM D# uses LDS padding: 256B of row data (pad_interval=5: 64 dwords),
  // then 96B pad (pad_amount=23: 24 dwords) => 352B row pitch = PCOLS halves.
#if HAVE_TDM
  if (tid < 32) {                       // wave 0 issues one TDM op (EXEC ignored)
    unsigned lds_dst = (unsigned)(uintptr_t)(void*)PATCH + (HR * PCOLS + HC) * 2;
    unsigned long long ga = (unsigned long long)(uintptr_t)src;
    v4u g0;
    g0[0] = 1u;                                         // count=1, user mode
    g0[1] = lds_dst;                                    // lds_addr [63:32]
    g0[2] = (unsigned)(ga & 0xFFFFFFFFu);               // global_addr lo
    g0[3] = (unsigned)((ga >> 32) & 0x1FFFFFFu)         // global_addr [56:32]
          | (2u << 30);                                 // type=2 ("image")
    v8i g1;
    g1[0] = (1 << 16)        // data_size = 1 -> 2 bytes
          | (1 << 20)        // pad_enable
          | (5 << 22)        // pad_interval: 64 dwords (256B)
          | (23 << 25);      // pad_amount: 24 dwords (96B)
    g1[1] = (HW & 0xFFFF) << 16;                        // tensor_dim0 lo16 @ [63:48]
    g1[2] = (HW >> 16) | ((HW & 0xFFFF) << 16);         // dim0 hi16, dim1 lo16
    g1[3] = (HW >> 16) | (PP << 16);                    // dim1 hi16, tile_dim0=128
    g1[4] = PP;                                         // tile_dim1=128, tile_dim2=0
    g1[5] = HW;                                         // tensor_dim0_stride lo32
    g1[6] = 0;                                          // stride hi16, dim1_stride lo16
    g1[7] = 0;
    v4i gz = {0, 0, 0, 0};                              // 2-D tensor: groups 2/3 unused
#if __clang_major__ >= 23
    v8i gz8 = {0, 0, 0, 0, 0, 0, 0, 0};
    __builtin_amdgcn_tensor_load_to_lds(g0, g1, gz, gz, gz8, 0);
#else
    __builtin_amdgcn_tensor_load_to_lds(g0, g1, gz, gz, 0);
#endif
    __builtin_amdgcn_s_wait_tensorcnt(0);
  }
#else
  for (int i = tid; i < PP * 16; i += 256) {            // fallback: manual copy
    int row = i >> 4;
    int q   = i & 15;                                   // 8 halves per step
    *(v8h*)(PATCH + (row + HR) * PCOLS + HC + q * 8) =
        *(const v8h*)(src + (size_t)row * HW + q * 8);
  }
#endif
  __syncthreads();

  // ---- phase 2: WMMA. wave w owns output tile-row w (16 rows), 8 tiles of 16 cols
  const int wave = tid >> 5;
  const int lane = tid & 31;
  const int m    = lane & 15;
  const bool lo  = lane < 16;
  const int y0   = wave * 16;
  const int jA0  = lo ? 0 : 8;     // A-fragment K offsets per ISA 16-bit A layout

  v8f acc[8] = {};

  for (int ky = 0; ky < 31; ++ky) {
    // B fragments: lane holds column N=lane&15, 16 K-halves in the same order as A
    const _Float16* bcol = TOEP + ky * 1024 + m;
    v16h bf0, bf1;
#pragma unroll
    for (int e = 0; e < 8; ++e) {
      bf0[e] = bcol[(jA0 + e) * 16]; bf0[8 + e] = bcol[(16 + jA0 + e) * 16];
      bf1[e] = bcol[(32 + jA0 + e) * 16]; bf1[8 + e] = bcol[(48 + jA0 + e) * 16];
    }

    const _Float16* arow = PATCH + (y0 + m + ky) * PCOLS;
#pragma unroll
    for (int tx = 0; tx < 8; ++tx) {
      const int x0 = tx * 16;
      v8h a0a = *(const v8h*)(arow + x0 + jA0);
      v8h a0b = *(const v8h*)(arow + x0 + jA0 + 16);
      v8h a1a = *(const v8h*)(arow + x0 + 32 + jA0);
      v8h a1b = *(const v8h*)(arow + x0 + 32 + jA0 + 16);
      v16h af0, af1;
#pragma unroll
      for (int e = 0; e < 8; ++e) {
        af0[e] = a0a[e]; af0[8 + e] = a0b[e];
        af1[e] = a1a[e]; af1[8 + e] = a1b[e];
      }
      acc[tx] = __builtin_amdgcn_wmma_f32_16x16x32_f16(
          false, af0, false, bf0, (short)0, acc[tx], false, false);
      acc[tx] = __builtin_amdgcn_wmma_f32_16x16x32_f16(
          false, af1, false, bf1, (short)0, acc[tx], false, false);
    }
  }

  // ---- epilogue: Hann window + overlap-add (C/D layout: VGPR r -> M=r(+8), N=lane&15)
#pragma unroll
  for (int tx = 0; tx < 8; ++tx) {
    const int px = tx * 16 + m;
    const float wx = HANN[px];
#pragma unroll
    for (int r = 0; r < 8; ++r) {
      const int py = y0 + r + (lo ? 0 : 8);
      const float v = acc[tx][r] * HANN[py] * wx;
      atomicAdd(out + ((size_t)plane * HW + (r0 + py)) * HW + (c0 + px), v);
    }
  }
}

// --------------------------- divide by analytic separable Hann weight-sum map
__device__ __forceinline__ float hann_at(int p) {
  return 0.5f * (1.0f - __cosf(6.28318530717958647692f * (float)p / 128.0f));
}

extern "C" __global__ void sv_norm(float* __restrict__ out, int ntot) {
  int i = blockIdx.x * blockDim.x + threadIdx.x;
  int stride = gridDim.x * blockDim.x;
  for (; i < ntot; i += stride) {
    int xg = i % HW;
    int yg = (i / HW) % HW;
    float wy = 0.0f, wx = 0.0f;
    {
      int ihh = yg >> 6;
      if (ihh <= 10) wy += hann_at(yg - (ihh << 6));
      int ihl = ihh - 1;
      if (ihl >= 0 && ihl <= 10) wy += hann_at(yg - (ihl << 6));
    }
    {
      int iwh = xg >> 6;
      if (iwh <= 10) wx += hann_at(xg - (iwh << 6));
      int iwl = iwh - 1;
      if (iwl >= 0 && iwl <= 10) wx += hann_at(xg - (iwl << 6));
    }
    out[i] = out[i] / (wy * wx + 1e-8f);
  }
}

// ---------------------------------------------------------------- launch glue
extern "C" void kernel_launch(void* const* d_in, const int* in_sizes, int n_in,
                              void* d_out, int out_size, void* d_ws, size_t ws_size,
                              hipStream_t stream) {
  const float* x     = (const float*)d_in[0];   // (2,3,768,768)
  const float* W1    = (const float*)d_in[1];   // (2,64)
  const float* b1    = (const float*)d_in[2];   // (64)
  const float* W2    = (const float*)d_in[3];   // (64,15)
  const float* b2    = (const float*)d_in[4];   // (15)
  const float* basis = (const float*)d_in[5];   // (15,3,31,31)
  float* out = (float*)d_out;

  float*    coeffs = (float*)d_ws;                       // 242*16 f32
  float*    kern   = (float*)((char*)d_ws + 16384);      // 242*3*961 f32 (~2.8 MB)
  _Float16* xh     = (_Float16*)((char*)d_ws + (3u << 20)); // f16 image (~7.1 MB)

  const int ntot = NB * NC * HW * HW;                    // 3,538,944

  sv_zero  <<<2048, 256, 0, stream>>>(out, ntot);
  sv_cast  <<<2048, 256, 0, stream>>>(x, xh, ntot / 4);
  sv_coeffs<<<1,    256, 0, stream>>>(W1, b1, W2, b2, coeffs);
  sv_kern  <<<NTOT * NC, 256, 0, stream>>>(coeffs, basis, kern);
  sv_conv  <<<NTOT * NC, 256, LDS_TOTAL_BYTES, stream>>>(xh, kern, out);
  sv_norm  <<<2048, 256, 0, stream>>>(out, ntot);
}
